// SuperPointMatching_14860586844160
// MI455X (gfx1250) — matrix operand order repaired and checked by reading, not compiled, not run on hardware
//
#include <hip/hip_runtime.h>
#include <hip/hip_bf16.h>

typedef __attribute__((ext_vector_type(16))) _Float16 v16h;
typedef __attribute__((ext_vector_type(8)))  _Float16 v8h;
typedef __attribute__((ext_vector_type(8)))  float    v8f;

constexpr int Mdim = 8192;
constexpr int Ndim = 8192;
constexpr int Ddim = 256;
constexpr int NCT  = Ndim / 16;   // 512 column tiles
constexpr int KST  = Ddim / 32;   // 8 WMMA K-steps per tile
constexpr int TOPK = 256;
constexpr int CAP  = 65536;       // candidate buffer capacity
constexpr int NBIN = 65536;       // float-bit histogram bins

// ---------------------------------------------------------------------------
// A-fragment loader: 16-bit A matrix 16x32, contiguous k order (B loaded the same way).
// lane<16: M=lane, kbase=0; lane>=16: same rows, kbase=16.
// v16h element e -> K = kbase + e
// => two contiguous 8-half (16B) chunks per lane.
// ---------------------------------------------------------------------------
__device__ __forceinline__ void load_A_frags(v16h af[KST],
                                             const _Float16* __restrict__ row,
                                             int half_sel) {
#pragma unroll
  for (int k0 = 0; k0 < KST; ++k0) {
    const _Float16* p = row + k0 * 32 + half_sel * 16;
    v8h lo = *(const v8h*)(p);
    v8h hi = *(const v8h*)(p + 8);
    v16h a;
#pragma unroll
    for (int e = 0; e < 8; ++e) { a[e] = lo[e]; a[e + 8] = hi[e]; }
    af[k0] = a;
  }
}

// ---------------------------------------------------------------------------
// Two 16x16 f32 dot tiles (row tiles i0, i0+16 vs one col tile) over K=256.
// Each B fragment (one contiguous 32B chunk of the src row per lane) feeds
// TWO v_wmma_f32_16x16x32_f16 ops -> 2x arithmetic intensity per B byte.
// ---------------------------------------------------------------------------
__device__ __forceinline__ void tile_dot2(const v16h af0[KST],
                                          const v16h af1[KST],
                                          const _Float16* __restrict__ brow,
                                          int half_sel, v8f& c0, v8f& c1) {
  v8f a0 = {}, a1 = {};
#pragma unroll
  for (int k0 = 0; k0 < KST; ++k0) {
    v16h b = *(const v16h*)(brow + k0 * 32 + half_sel * 16);
    a0 = __builtin_amdgcn_wmma_f32_16x16x32_f16(false, af0[k0], false, b,
                                                (short)0, a0, false, false);
    a1 = __builtin_amdgcn_wmma_f32_16x16x32_f16(false, af1[k0], false, b,
                                                (short)0, a1, false, false);
  }
  c0 = a0;
  c1 = a1;
}

// ---------------------------------------------------------------------------
// f32 -> f16 conversion
// ---------------------------------------------------------------------------
__global__ void cvt_kernel(const float* __restrict__ in,
                           _Float16* __restrict__ out, int n) {
  for (int i = blockIdx.x * blockDim.x + threadIdx.x; i < n;
       i += gridDim.x * blockDim.x)
    out[i] = (_Float16)in[i];
}

// ---------------------------------------------------------------------------
// Pass 1/2: sums[i] += sum_j exp(2*dot(A_i,B_j) - 2).
// Wave handles 32 rows (2 row tiles) x 1024 cols (64 col tiles).
// grid = (8, 32), block = 256 (8 waves).
// ---------------------------------------------------------------------------
__global__ void __launch_bounds__(256) exp_rowsum_kernel(
    const _Float16* __restrict__ A, const _Float16* __restrict__ B,
    float* __restrict__ sums) {
  const int tid = threadIdx.x;
  const int lane = tid & 31, wave = tid >> 5;
  const int half_sel = lane >> 4, l15 = lane & 15;
  const int i0 = (blockIdx.y * 8 + wave) * 32;  // first of 2 row tiles
  const int j_begin = blockIdx.x * 1024;

  v16h af0[KST], af1[KST];
  load_A_frags(af0, A + (size_t)(i0 + l15) * Ddim, half_sel);
  load_A_frags(af1, A + (size_t)(i0 + 16 + l15) * Ddim, half_sel);

  float rsum0[8], rsum1[8];
#pragma unroll
  for (int r = 0; r < 8; ++r) { rsum0[r] = 0.f; rsum1[r] = 0.f; }

  for (int jt = 0; jt < 64; ++jt) {
    const int j0 = j_begin + jt * 16;
    // start pulling next tile's B rows toward WGP$ early
    const int jn = (jt < 63) ? (j0 + 16) : j0;
    __builtin_prefetch(B + (size_t)(jn + l15) * Ddim, 0, 1);

    v8f c0, c1;
    tile_dot2(af0, af1, B + (size_t)(j0 + l15) * Ddim, half_sel, c0, c1);
#pragma unroll
    for (int r = 0; r < 8; ++r) {
      rsum0[r] += __expf(fmaf(2.f, c0[r], -2.f));
      rsum1[r] += __expf(fmaf(2.f, c1[r], -2.f));
    }
  }
  // reduce over the 16 lanes of each half-wave (xor masks stay within halves)
#pragma unroll
  for (int r = 0; r < 8; ++r) {
    float v = rsum0[r];
    v += __shfl_xor(v, 8, 32);
    v += __shfl_xor(v, 4, 32);
    v += __shfl_xor(v, 2, 32);
    v += __shfl_xor(v, 1, 32);
    rsum0[r] = v;
    float w = rsum1[r];
    w += __shfl_xor(w, 8, 32);
    w += __shfl_xor(w, 4, 32);
    w += __shfl_xor(w, 2, 32);
    w += __shfl_xor(w, 1, 32);
    rsum1[r] = w;
  }
  if (l15 == 0) {
    const int rowbase = i0 + half_sel * 8;
#pragma unroll
    for (int r = 0; r < 8; ++r) {
      atomicAdd(&sums[rowbase + r], rsum0[r]);
      atomicAdd(&sums[rowbase + 16 + r], rsum1[r]);
    }
  }
}

// ---------------------------------------------------------------------------
// Pass 3: per-(row, 16-col segment) max of v = E^2/(rowsum*colsum).
// Wave handles 2 row tiles x 8 col tiles; grid = (64, 32), block = 256.
// ---------------------------------------------------------------------------
__global__ void __launch_bounds__(256) segmax_kernel(
    const _Float16* __restrict__ A, const _Float16* __restrict__ B,
    const float* __restrict__ rowsum, const float* __restrict__ colsum,
    float* __restrict__ segmax) {
  const int tid = threadIdx.x;
  const int lane = tid & 31, wave = tid >> 5;
  const int half_sel = lane >> 4, l15 = lane & 15;
  const int i0 = (blockIdx.y * 8 + wave) * 32;

  v16h af0[KST], af1[KST];
  load_A_frags(af0, A + (size_t)(i0 + l15) * Ddim, half_sel);
  load_A_frags(af1, A + (size_t)(i0 + 16 + l15) * Ddim, half_sel);

  float rs0[8], rs1[8];
#pragma unroll
  for (int r = 0; r < 8; ++r) {
    rs0[r] = rowsum[i0 + half_sel * 8 + r];
    rs1[r] = rowsum[i0 + 16 + half_sel * 8 + r];
  }

  for (int jj = 0; jj < 8; ++jj) {
    const int jt = blockIdx.x * 8 + jj;
    const int j0 = jt * 16;
    const int jn = (jj < 7) ? (j0 + 16) : j0;
    __builtin_prefetch(B + (size_t)(jn + l15) * Ddim, 0, 1);
    const float cs = colsum[j0 + l15];

    v8f c0, c1;
    tile_dot2(af0, af1, B + (size_t)(j0 + l15) * Ddim, half_sel, c0, c1);
#pragma unroll
    for (int r = 0; r < 8; ++r) {
      float E0 = __expf(fmaf(2.f, c0[r], -2.f));
      float v0 = (E0 * E0) / (rs0[r] * cs);
      v0 = fmaxf(v0, __shfl_xor(v0, 8, 32));
      v0 = fmaxf(v0, __shfl_xor(v0, 4, 32));
      v0 = fmaxf(v0, __shfl_xor(v0, 2, 32));
      v0 = fmaxf(v0, __shfl_xor(v0, 1, 32));
      float E1 = __expf(fmaf(2.f, c1[r], -2.f));
      float v1 = (E1 * E1) / (rs1[r] * cs);
      v1 = fmaxf(v1, __shfl_xor(v1, 8, 32));
      v1 = fmaxf(v1, __shfl_xor(v1, 4, 32));
      v1 = fmaxf(v1, __shfl_xor(v1, 2, 32));
      v1 = fmaxf(v1, __shfl_xor(v1, 1, 32));
      if (l15 == 0) {
        segmax[(size_t)(i0 + half_sel * 8 + r) * NCT + jt] = v0;
        segmax[(size_t)(i0 + 16 + half_sel * 8 + r) * NCT + jt] = v1;
      }
    }
  }
}

// ---------------------------------------------------------------------------
// Histogram of segment maxima (positive floats: bits monotonic).
// ---------------------------------------------------------------------------
__global__ void hist_kernel(const float* __restrict__ segmax,
                            unsigned* __restrict__ hist) {
  int i = blockIdx.x * blockDim.x + threadIdx.x;
  if (i < Mdim * NCT) {
    unsigned b = __float_as_uint(segmax[i]) >> 16;
    atomicAdd(&hist[b], 1u);
  }
}

// tau = floor of the bin containing the 256th-largest segment max.
// tau <= true 256th-largest matrix value, so collecting all values >= tau
// yields a superset of the global top-256.
__global__ void thresh_kernel(const unsigned* __restrict__ hist,
                              float* __restrict__ tau_p) {
  if (threadIdx.x == 0 && blockIdx.x == 0) {
    unsigned acc = 0;
    int b = NBIN - 1;
    for (; b > 0; --b) {
      acc += hist[b];
      if (acc >= (unsigned)TOPK) break;
    }
    *tau_p = __uint_as_float(((unsigned)b) << 16);
  }
}

// ---------------------------------------------------------------------------
// Pass 4: recompute v, push (v, linear index) for v >= tau.
// Same tiling as segmax. Identical arithmetic => identical values.
// ---------------------------------------------------------------------------
__global__ void __launch_bounds__(256) collect_kernel(
    const _Float16* __restrict__ A, const _Float16* __restrict__ B,
    const float* __restrict__ rowsum, const float* __restrict__ colsum,
    const float* __restrict__ tau_p, float* __restrict__ cand_val,
    int* __restrict__ cand_idx, unsigned* __restrict__ counter) {
  const int tid = threadIdx.x;
  const int lane = tid & 31, wave = tid >> 5;
  const int half_sel = lane >> 4, l15 = lane & 15;
  const int i0 = (blockIdx.y * 8 + wave) * 32;
  const float tau = *tau_p;

  v16h af0[KST], af1[KST];
  load_A_frags(af0, A + (size_t)(i0 + l15) * Ddim, half_sel);
  load_A_frags(af1, A + (size_t)(i0 + 16 + l15) * Ddim, half_sel);

  float rs0[8], rs1[8];
#pragma unroll
  for (int r = 0; r < 8; ++r) {
    rs0[r] = rowsum[i0 + half_sel * 8 + r];
    rs1[r] = rowsum[i0 + 16 + half_sel * 8 + r];
  }

  for (int jj = 0; jj < 8; ++jj) {
    const int jt = blockIdx.x * 8 + jj;
    const int j0 = jt * 16;
    const int jn = (jj < 7) ? (j0 + 16) : j0;
    __builtin_prefetch(B + (size_t)(jn + l15) * Ddim, 0, 1);
    const float cs = colsum[j0 + l15];

    v8f c0, c1;
    tile_dot2(af0, af1, B + (size_t)(j0 + l15) * Ddim, half_sel, c0, c1);
#pragma unroll
    for (int r = 0; r < 8; ++r) {
      float E0 = __expf(fmaf(2.f, c0[r], -2.f));
      float v0 = (E0 * E0) / (rs0[r] * cs);
      if (v0 >= tau) {
        unsigned pos = atomicAdd(counter, 1u);
        if (pos < (unsigned)CAP) {
          cand_val[pos] = v0;
          cand_idx[pos] = (i0 + half_sel * 8 + r) * Ndim + (j0 + l15);
        }
      }
      float E1 = __expf(fmaf(2.f, c1[r], -2.f));
      float v1 = (E1 * E1) / (rs1[r] * cs);
      if (v1 >= tau) {
        unsigned pos = atomicAdd(counter, 1u);
        if (pos < (unsigned)CAP) {
          cand_val[pos] = v1;
          cand_idx[pos] = (i0 + 16 + half_sel * 8 + r) * Ndim + (j0 + l15);
        }
      }
    }
  }
}

// ---------------------------------------------------------------------------
// Exact, descending, tie-stable top-256 among collected candidates.
// Single block; 256 rounds of block-wide argmax with -1 marking.
// ---------------------------------------------------------------------------
__global__ void __launch_bounds__(256) final_topk_kernel(
    float* __restrict__ cand_val, const int* __restrict__ cand_idx,
    const unsigned* __restrict__ counter, float* __restrict__ out) {
  __shared__ float sv[256];
  __shared__ int sp[256];
  const int t = threadIdx.x;
  unsigned cu = *counter;
  int n = (cu > (unsigned)CAP) ? CAP : (int)cu;

  for (int iter = 0; iter < TOPK; ++iter) {
    float bv = -1.f;
    int bp = -1, bidx = 0x7fffffff;
    for (int i = t; i < n; i += 256) {
      float v = cand_val[i];
      int idx = cand_idx[i];
      if (v > bv || (v == bv && idx < bidx)) { bv = v; bp = i; bidx = idx; }
    }
    sv[t] = bv;
    sp[t] = bp;
    __syncthreads();
    for (int s = 128; s > 0; s >>= 1) {
      if (t < s) {
        float ov = sv[t + s];
        int op = sp[t + s];
        int oidx = (op >= 0) ? cand_idx[op] : 0x7fffffff;
        int midx = (sp[t] >= 0) ? cand_idx[sp[t]] : 0x7fffffff;
        if (ov > sv[t] || (ov == sv[t] && oidx < midx)) { sv[t] = ov; sp[t] = op; }
      }
      __syncthreads();
    }
    if (t == 0) {
      int p = sp[0];
      if (p >= 0) {
        int idx = cand_idx[p];
        out[iter]            = (float)(idx / Ndim);
        out[TOPK + iter]     = (float)(idx % Ndim);
        out[2 * TOPK + iter] = sv[0];
        cand_val[p] = -1.f;  // consume
      } else {
        out[iter] = 0.f;
        out[TOPK + iter] = 0.f;
        out[2 * TOPK + iter] = 0.f;
      }
    }
    __syncthreads();
  }
}

// ---------------------------------------------------------------------------
extern "C" void kernel_launch(void* const* d_in, const int* in_sizes, int n_in,
                              void* d_out, int out_size, void* d_ws,
                              size_t ws_size, hipStream_t stream) {
  (void)in_sizes; (void)n_in; (void)out_size; (void)ws_size;
  const float* ref = (const float*)d_in[0];
  const float* src = (const float*)d_in[1];
  float* out = (float*)d_out;

  char* ws = (char*)d_ws;
  size_t off = 0;
  _Float16* refh = (_Float16*)(ws + off); off += (size_t)Mdim * Ddim * 2;
  _Float16* srch = (_Float16*)(ws + off); off += (size_t)Ndim * Ddim * 2;
  float* rowsum = (float*)(ws + off);     off += (size_t)Mdim * 4;
  float* colsum = (float*)(ws + off);     off += (size_t)Ndim * 4;
  unsigned* hist = (unsigned*)(ws + off); off += (size_t)NBIN * 4;
  unsigned* counter = (unsigned*)(ws + off); off += 256;
  float* tau = (float*)(ws + off);        off += 256;
  float* cand_val = (float*)(ws + off);   off += (size_t)CAP * 4;
  int* cand_idx = (int*)(ws + off);       off += (size_t)CAP * 4;
  float* segmax = (float*)(ws + off);     off += (size_t)Mdim * NCT * 4;

  // Deterministic per-call init (graph-capture legal).
  hipMemsetAsync(rowsum, 0, (size_t)(Mdim + Ndim) * 4, stream);  // row+col sums
  hipMemsetAsync(hist, 0, (size_t)NBIN * 4 + 256, stream);       // hist+counter

  cvt_kernel<<<2048, 256, 0, stream>>>(ref, refh, Mdim * Ddim);
  cvt_kernel<<<2048, 256, 0, stream>>>(src, srch, Ndim * Ddim);

  dim3 gsum(8, 32);  // 8 col-chunks x 32 blocks of 8 waves (32 rows each)
  exp_rowsum_kernel<<<gsum, 256, 0, stream>>>(refh, srch, rowsum);
  exp_rowsum_kernel<<<gsum, 256, 0, stream>>>(srch, refh, colsum);

  dim3 gmat(NCT / 8, Mdim / 32 / 8);  // (64, 32)
  segmax_kernel<<<gmat, 256, 0, stream>>>(refh, srch, rowsum, colsum, segmax);

  hist_kernel<<<(Mdim * NCT) / 256, 256, 0, stream>>>(segmax, hist);
  thresh_kernel<<<1, 32, 0, stream>>>(hist, tau);

  collect_kernel<<<gmat, 256, 0, stream>>>(refh, srch, rowsum, colsum, tau,
                                           cand_val, cand_idx, counter);
  final_topk_kernel<<<1, 256, 0, stream>>>(cand_val, cand_idx, counter, out);
}
